// FastFFlayer_32873679684249
// MI455X (gfx1250) — compile-verified, matrix operand order checked by reading, not run on hardware
//
#include <hip/hip_runtime.h>
#include <hip/hip_bf16.h>

typedef __bf16 bf16;
typedef __attribute__((ext_vector_type(16))) __bf16 bf16x16;
typedef __attribute__((ext_vector_type(8)))  __bf16 bf16x8;
typedef __attribute__((ext_vector_type(8)))  float  floatx8;

#define SLEN 1024
#define BSZ  16
#define NH   8
#define DH   32
#define IND  256
#define PROJ 776            // NH*(3*DH+1)
#define QSTRIDE 800         // padded qkvb row stride (25 * 32)
#define M_ROWS (SLEN*BSZ)   // 16384
#define OUT_ELEMS (SLEN*BSZ*IND) // 4194304

// ---------------------------------------------------------------------------
// lane-broadcast via v_readlane_b32 (uniform result -> SGPR, no DS traffic)
// ---------------------------------------------------------------------------
__device__ inline float bcast(float v, int lane) {
    return __builtin_bit_cast(float,
        __builtin_amdgcn_readlane(__builtin_bit_cast(int, v), lane));
}

// ---------------------------------------------------------------------------
// Kernel 1: LayerNorm -> bf16  (one block of 256 threads per row)
// ---------------------------------------------------------------------------
__global__ void ln_to_bf16(const float* __restrict__ x,
                           const float* __restrict__ gamma,
                           const float* __restrict__ beta,
                           bf16* __restrict__ out) {
    const int m   = blockIdx.x;
    const int tid = threadIdx.x;
    const float v = x[(size_t)m * IND + tid];

    float s1 = v, s2 = v * v;
    #pragma unroll
    for (int off = 16; off >= 1; off >>= 1) {
        s1 += __shfl_xor(s1, off);
        s2 += __shfl_xor(s2, off);
    }
    __shared__ float a1[8], a2[8];
    const int w = tid >> 5, l = tid & 31;
    if (l == 0) { a1[w] = s1; a2[w] = s2; }
    __syncthreads();
    float t1 = 0.f, t2 = 0.f;
    #pragma unroll
    for (int i = 0; i < 8; i++) { t1 += a1[i]; t2 += a2[i]; }
    const float mu  = t1 * (1.0f / IND);
    const float var = t2 * (1.0f / IND) - mu * mu;
    const float r   = rsqrtf(var + 1e-5f);
    out[(size_t)m * IND + tid] = (bf16)((v - mu) * r * gamma[tid] + beta[tid]);
}

// ---------------------------------------------------------------------------
// Kernel 2: f32 -> bf16 conversion, zero-padded to n_out elements
// ---------------------------------------------------------------------------
__global__ void cvt_bf16_pad(const float* __restrict__ in, bf16* __restrict__ out,
                             int n_in, int n_out) {
    int i = blockIdx.x * blockDim.x + threadIdx.x;
    if (i < n_out) out[i] = (i < n_in) ? (bf16)in[i] : (bf16)0.0f;
}

// ---------------------------------------------------------------------------
// WMMA fragment load (ISA 7.12.2 16-bit A/B layout, wave32)
// uniform base pointer + 32-bit per-lane offset -> saddr+voffset form
// ---------------------------------------------------------------------------
__device__ inline bf16x16 ld_frag(const bf16* __restrict__ base, int off) {
    bf16x8 lo = *(const bf16x8*)(base + off);
    bf16x8 hi = *(const bf16x8*)(base + off + 16);
    return __builtin_shufflevector(lo, hi, 0,1,2,3,4,5,6,7,8,9,10,11,12,13,14,15);
}

// ---------------------------------------------------------------------------
// Kernel 3: qkvb = LN(x) @ W_slow^T -> f32 [16384, 800(padded)]
// One wave computes a 64x32 macro-tile (4x2 grid of 16x16 WMMA tiles):
// 12 b128 loads per k-chunk feed 8 v_wmma; ~112 live VGPRs -> no spills.
// ---------------------------------------------------------------------------
__global__ void gemm_qkvb(const bf16* __restrict__ A,
                          const bf16* __restrict__ Wq,
                          float* __restrict__ C) {
    const int lane  = threadIdx.x;
    const int l15   = lane & 15;
    const int khalf = (lane >> 4) << 3;          // 0 or 8
    const int n0    = blockIdx.x * 32;           // 25 groups -> 800 cols
    const int m0    = blockIdx.y * 64;

    const bf16* Ab = A + (size_t)m0 * IND;       // uniform base
    int aoff[4], boff[2];
    #pragma unroll
    for (int i = 0; i < 4; i++) aoff[i] = (i * 16 + l15) * IND + khalf;
    #pragma unroll
    for (int j = 0; j < 2; j++) boff[j] = (n0 + j * 16 + l15) * IND + khalf;

    floatx8 c[8] = {};
    #pragma unroll
    for (int kc = 0; kc < IND; kc += 32) {
        bf16x16 a[4], bm[2];
        #pragma unroll
        for (int j = 0; j < 2; j++) bm[j] = ld_frag(Wq, boff[j] + kc);
        #pragma unroll
        for (int i = 0; i < 4; i++) a[i] = ld_frag(Ab, aoff[i] + kc);
        #pragma unroll
        for (int i = 0; i < 4; i++)
            #pragma unroll
            for (int j = 0; j < 2; j++)
                c[i * 2 + j] = __builtin_amdgcn_wmma_f32_16x16x32_bf16(
                    false, a[i], false, bm[j], (short)0, c[i * 2 + j], false, false);
    }

    float* Cb = C + (size_t)m0 * QSTRIDE;        // uniform base
    #pragma unroll
    for (int i = 0; i < 4; i++) {
        const int rb = i * 16 + ((lane >> 4) << 3);
        #pragma unroll
        for (int j = 0; j < 2; j++) {
            const int col = n0 + j * 16 + l15;
            #pragma unroll
            for (int r = 0; r < 8; r++)
                Cb[(rb + r) * QSTRIDE + col] = c[i * 2 + j][r];
        }
    }
}

// ---------------------------------------------------------------------------
// Kernel 4: delta-rule recurrence. One wave32 per (b,h); lane i holds W[i,:]
// in 32 VGPRs. Broadcasts via v_readlane (SGPR), fused matvecs:
//   o_t = W_old q + delta * (k . q),  then W += delta k^T (deferred update).
// Next step's q/k/v/beta loads issued before current step's math.
// ---------------------------------------------------------------------------
__global__ void recurrence(const float* __restrict__ qkvb,
                           const float* __restrict__ state,
                           bf16*  __restrict__ oseq,
                           float* __restrict__ fw_out) {
    const int bh   = blockIdx.x;       // 0..127  (b*NH + h)
    const int b    = bh >> 3;
    const int h    = bh & 7;
    const int lane = threadIdx.x;      // row index i

    float w[DH];
    {
        const float* st = state + (size_t)bh * (DH * DH) + (size_t)lane * DH;
        #pragma unroll
        for (int j = 0; j < DH; j++) w[j] = st[j];
    }

    // preload t = 0
    size_t base = (size_t)b * QSTRIDE + h * (3 * DH + 1);
    float q  = qkvb[base + lane];
    float k  = qkvb[base + DH + lane];
    float v  = qkvb[base + 2 * DH + lane];
    float br = qkvb[base + 3 * DH];

    for (int t = 0; t < SLEN; t++) {
        // issue next step's loads early (overlap with this step's chain)
        const int tn = (t + 1 < SLEN) ? t + 1 : SLEN - 1;
        const size_t nb = ((size_t)(tn * BSZ + b)) * QSTRIDE + h * (3 * DH + 1);
        const float qn  = qkvb[nb + lane];
        const float kn  = qkvb[nb + DH + lane];
        const float vn  = qkvb[nb + 2 * DH + lane];
        const float brn = qkvb[nb + 3 * DH];

        const float beta = 1.0f / (1.0f + __expf(-br));
        float qe = (q > 0.f) ? q + 1.f : __expf(q);
        float ke = (k > 0.f) ? k + 1.f : __expf(k);
        float qs = qe, ks = ke;
        #pragma unroll
        for (int off = 16; off >= 1; off >>= 1) {
            qs += __shfl_xor(qs, off);
            ks += __shfl_xor(ks, off);
        }
        qe = qe / (qs + 1e-5f);
        ke = ke / (ks + 1e-5f);

        // dot(k, q) wave reduction
        float kq = ke * qe;
        #pragma unroll
        for (int off = 16; off >= 1; off >>= 1) kq += __shfl_xor(kq, off);

        // fused matvecs against pre-update W (two accumulator pairs)
        float ksc[DH];
        float vo0 = 0.f, vo1 = 0.f, oq0 = 0.f, oq1 = 0.f;
        #pragma unroll
        for (int j = 0; j < DH; j += 2) {
            ksc[j]     = bcast(ke, j);
            ksc[j + 1] = bcast(ke, j + 1);
            const float qj0 = bcast(qe, j);
            const float qj1 = bcast(qe, j + 1);
            vo0 = fmaf(w[j],     ksc[j],     vo0);
            oq0 = fmaf(w[j],     qj0,        oq0);
            vo1 = fmaf(w[j + 1], ksc[j + 1], vo1);
            oq1 = fmaf(w[j + 1], qj1,        oq1);
        }
        const float delta = beta * (v - (vo0 + vo1));
        const float o     = fmaf(delta, kq, oq0 + oq1);

        #pragma unroll
        for (int j = 0; j < DH; j++) w[j] = fmaf(delta, ksc[j], w[j]);

        oseq[(size_t)(t * BSZ + b) * IND + h * DH + lane] = (bf16)o;

        q = qn; k = kn; v = vn; br = brn;
    }

    float* fw = fw_out + (size_t)bh * (DH * DH) + (size_t)lane * DH;
    #pragma unroll
    for (int j = 0; j < DH; j++) fw[j] = w[j];
}

// ---------------------------------------------------------------------------
// Kernel 5: out = x + o_seq @ W_out^T   (64x32 macro-tile per wave, N=256)
// ---------------------------------------------------------------------------
__global__ void gemm_out(const bf16* __restrict__ A,
                         const bf16* __restrict__ Wo,
                         const float* __restrict__ x,
                         float* __restrict__ out) {
    const int lane  = threadIdx.x;
    const int l15   = lane & 15;
    const int khalf = (lane >> 4) << 3;
    const int n0    = blockIdx.x * 32;           // 8 groups
    const int m0    = blockIdx.y * 64;

    const bf16* Ab = A + (size_t)m0 * IND;
    int aoff[4], boff[2];
    #pragma unroll
    for (int i = 0; i < 4; i++) aoff[i] = (i * 16 + l15) * IND + khalf;
    #pragma unroll
    for (int j = 0; j < 2; j++) boff[j] = (n0 + j * 16 + l15) * IND + khalf;

    floatx8 c[8] = {};
    #pragma unroll
    for (int kc = 0; kc < IND; kc += 32) {
        bf16x16 a[4], bm[2];
        #pragma unroll
        for (int j = 0; j < 2; j++) bm[j] = ld_frag(Wo, boff[j] + kc);
        #pragma unroll
        for (int i = 0; i < 4; i++) a[i] = ld_frag(Ab, aoff[i] + kc);
        #pragma unroll
        for (int i = 0; i < 4; i++)
            #pragma unroll
            for (int j = 0; j < 2; j++)
                c[i * 2 + j] = __builtin_amdgcn_wmma_f32_16x16x32_bf16(
                    false, a[i], false, bm[j], (short)0, c[i * 2 + j], false, false);
    }

    const float* xb = x   + (size_t)m0 * IND;
    float*      ob  = out + (size_t)m0 * IND;
    #pragma unroll
    for (int i = 0; i < 4; i++) {
        const int rb = i * 16 + ((lane >> 4) << 3);
        #pragma unroll
        for (int j = 0; j < 2; j++) {
            const int col = n0 + j * 16 + l15;
            #pragma unroll
            for (int r = 0; r < 8; r++) {
                const int idx = (rb + r) * IND + col;
                ob[idx] = xb[idx] + c[i * 2 + j][r];
            }
        }
    }
}

// ---------------------------------------------------------------------------
extern "C" void kernel_launch(void* const* d_in, const int* in_sizes, int n_in,
                              void* d_out, int out_size, void* d_ws, size_t ws_size,
                              hipStream_t stream) {
    const float* x       = (const float*)d_in[0];
    const float* state   = (const float*)d_in[1];
    const float* W_slow  = (const float*)d_in[2];
    const float* ln_g    = (const float*)d_in[3];
    const float* ln_b    = (const float*)d_in[4];
    const float* W_out   = (const float*)d_in[5];
    float* out = (float*)d_out;

    // workspace layout (bf16 region first, then f32 qkvb)
    bf16* ln_bf    = (bf16*)d_ws;                        // 16384*256
    bf16* wslow_bf = ln_bf    + (size_t)M_ROWS * IND;    // 800*256 (zero padded)
    bf16* wout_bf  = wslow_bf + (size_t)QSTRIDE * IND;   // 256*256
    bf16* oseq_bf  = wout_bf  + (size_t)IND * IND;       // 16384*256
    float* qkvb    = (float*)(oseq_bf + (size_t)M_ROWS * IND); // 16384*800

    // 1) LayerNorm -> bf16
    ln_to_bf16<<<M_ROWS, IND, 0, stream>>>(x, ln_g, ln_b, ln_bf);

    // 2) weight conversions (W_slow zero-padded to 800 rows)
    {
        int n1i = PROJ * IND, n1o = QSTRIDE * IND;
        cvt_bf16_pad<<<(n1o + 255) / 256, 256, 0, stream>>>(W_slow, wslow_bf, n1i, n1o);
        int n2 = IND * IND;
        cvt_bf16_pad<<<(n2 + 255) / 256, 256, 0, stream>>>(W_out, wout_bf, n2, n2);
    }

    // 3) projection GEMM (WMMA, 64x32 macro-tiles): qkvb[16384, 800]
    {
        dim3 grid(QSTRIDE / 32, M_ROWS / 64);   // 25 x 256
        gemm_qkvb<<<grid, 32, 0, stream>>>(ln_bf, wslow_bf, qkvb);
    }

    // 4) delta-rule recurrence: one wave per (b,h); fast weights -> d_out tail
    recurrence<<<BSZ * NH, 32, 0, stream>>>(qkvb, state, oseq_bf,
                                            out + OUT_ELEMS);

    // 5) output GEMM (WMMA, 64x32 macro-tiles) + residual
    {
        dim3 grid(IND / 32, M_ROWS / 64);       // 8 x 256
        gemm_out<<<grid, 32, 0, stream>>>(oseq_bf, wout_bf, x, out);
    }
}
